// Encoder_22720376996265
// MI455X (gfx1250) — matrix operand order repaired and checked by reading, not compiled, not run on hardware
//
// CDNA5 / gfx1250 implementation of the GCN+attention encoder.
//
// Key observations (see analysis): x is never updated across GCN layers, so the
// node path (gate/agg/nU/nV/bn_x) is dead code. Only the edge path survives:
//   e0 -> [GEMM 64x64 + Vx_i + Vx_j] -> BN stats -> BN+relu residual -> layer2
//   -> gather 256 edges -> W_el -> attention.
// Memory-bound (~200MB of e-tensor traffic vs ~4.3 GFLOP), so intermediates are
// bf16 and the per-edge row GEMMs use v_wmma_f32_16x16x32_bf16.
//
// Workspace requirement: ~68 MB (two 32MB bf16 e-buffers + small scratch).

#include <hip/hip_runtime.h>
#include <hip/hip_bf16.h>
#include <math.h>

typedef __attribute__((ext_vector_type(16))) __bf16 v16bf;
typedef __attribute__((ext_vector_type(8)))  __bf16 v8bf;
typedef __attribute__((ext_vector_type(8)))  float  v8f;

#define HID    64
#define NN     256
#define NB     4
#define LLEN   64
#define EODX   28
#define ROWS   (NB*NN*NN)      /* 262144 edge rows */
#define TILES  (ROWS/16)       /* 16384 */
#define GCN_BLOCKS (TILES/8)   /* 2048, 8 waves/block, 1 tile/wave */

static __device__ __forceinline__ float  bf2f(__bf16 x){ return (float)x; }
static __device__ __forceinline__ __bf16 f2bf(float x){ return (__bf16)x; }

// ---------------------------------------------------------------------------
// Pack the 64x64 f32 eU_w weights into WMMA B-fragment order (bf16).
// Fragment order: p = ((h*4+t)*32 + lane)*16 + e, value U[k][n] with
//   k = 32h + (lane>=16 ? 8 : 0) + (e<8 ? e : e+8),  n = 16t + (lane&15)
// so each lane's v16bf B-fragment is one contiguous 32B load.
// ---------------------------------------------------------------------------
__global__ __launch_bounds__(256) void k_prep_w(const float* __restrict__ U1,
                                                const float* __restrict__ U2,
                                                __bf16* __restrict__ U1pk,
                                                __bf16* __restrict__ U2pk)
{
    for (int p = threadIdx.x; p < 4096; p += 256) {
        const int e = p & 15;
        const int l = (p >> 4) & 31;
        const int t = (p >> 9) & 3;
        const int h = p >> 11;
        const int k = 32*h + ((l & 16) ? 8 : 0) + (e < 8 ? e : e + 8);
        const int n = 16*t + (l & 15);
        U1pk[p] = f2bf(U1[k*HID + n]);
        U2pk[p] = f2bf(U2[k*HID + n]);
    }
}

// ---------------------------------------------------------------------------
// VxB[layer][b,n,c] = (x @ eV_w)[c] + eV_b[c] + 0.5*eU_b[c],  x = node@W_node.
// (0.5*eU_b so that VxB[i]+VxB[j] carries the full eU_b bias.)
// ---------------------------------------------------------------------------
__global__ __launch_bounds__(64) void k_prep_vx(
    const float* __restrict__ f, const float* __restrict__ node_tab,
    const float* __restrict__ W_node,
    const float* __restrict__ eVw1, const float* __restrict__ eVb1, const float* __restrict__ eUb1,
    const float* __restrict__ eVw2, const float* __restrict__ eVb2, const float* __restrict__ eUb2,
    float* __restrict__ VxB1, float* __restrict__ VxB2)
{
    __shared__ float xs[HID];
    const int b = blockIdx.x >> 8;
    const int n = blockIdx.x & 255;
    const int c = threadIdx.x;
    const int t = (int)f[b*9];
    const float* nd = node_tab + ((size_t)(t*NN + n))*8 + 1;
    float x = 0.f;
    #pragma unroll
    for (int k = 0; k < 7; ++k) x += nd[k] * W_node[k*HID + c];
    xs[c] = x;
    __syncthreads();
    float v1 = eVb1[c] + 0.5f*eUb1[c];
    float v2 = eVb2[c] + 0.5f*eUb2[c];
    for (int k = 0; k < HID; ++k) {
        const float xv = xs[k];
        v1 += xv * eVw1[k*HID + c];
        v2 += xv * eVw2[k*HID + c];
    }
    VxB1[(size_t)blockIdx.x*HID + c] = v1;
    VxB2[(size_t)blockIdx.x*HID + c] = v2;
}

// ---------------------------------------------------------------------------
// Main GCN edge pass.  One wave handles a 16-row tile (b,i fixed, j=j0..j0+15):
//   LAYER==1: build e0 from edge_tab/W_eval/emb_edges[A], write e0 -> bufE
//   LAYER==2: e1 = e0 + relu(bn1(etmp1)) from bufE/bufT, write e1 -> bufE
// then etmp = tile @ Upk (WMMA bf16, K=64) + VxB[i] + VxB[j] -> bufT,
// plus deterministic per-block partial sums/sumsq for BN stats.
// ---------------------------------------------------------------------------
template<int LAYER>
__global__ __launch_bounds__(256) void k_gcn(
    const float* __restrict__ f,
    const float* __restrict__ edge_tab,
    const int*   __restrict__ A_tab,
    const float* __restrict__ W_eval,
    const float* __restrict__ emb_edges,
    const __bf16* __restrict__ Upk,
    const float* __restrict__ VxB,
    const float* __restrict__ bn_g,
    const float* __restrict__ bn_b,
    const float* __restrict__ musig,
    __bf16* __restrict__ bufE,
    __bf16* __restrict__ bufT,
    float*  __restrict__ partial)
{
    __shared__ __align__(16) __bf16 atile[8][16*HID];
    __shared__ __align__(16) __bf16 ctile[8][16*HID];
    __shared__ float sS[8][64];
    __shared__ float sQ[8][64];
    __shared__ float cst[256];

    const int wave = threadIdx.x >> 5;
    const int lane = threadIdx.x & 31;
    const int tIdx = blockIdx.x * 8 + wave;
    const int j0 = (tIdx & 15) << 4;
    const int i  = (tIdx >> 4) & 255;
    const int b  =  tIdx >> 12;

    if (LAYER == 1) {
        if (threadIdx.x < 128) cst[threadIdx.x] = W_eval[threadIdx.x];
        if (threadIdx.x < 96)  cst[128 + threadIdx.x] = emb_edges[threadIdx.x];
    } else {
        if (threadIdx.x < 64) {
            const int c = threadIdx.x;
            cst[c]       = bn_g[c];
            cst[64 + c]  = bn_b[c];
            cst[128 + c] = musig[c];        // mu1
            cst[192 + c] = musig[64 + c];   // rsqrt(var1+eps)
        }
    }
    __syncthreads();

    // -------- phase 1: stage 16x64 bf16 input tile in LDS ------------------
    const int m  = lane >> 1;
    const int cH = (lane & 1) << 5;   // 0 or 32
    __bf16* tile = atile[wave];

    if (LAYER == 1) {
        const int tsel = (int)f[b*9];
        const size_t eb = ((size_t)(tsel*NN + i))*NN + (size_t)(j0 + m);
        if (cH == 0) {
            const float* ep = edge_tab + eb*6 + 2;
            const float e0v = ep[0], e1v = ep[1], e2v = ep[2], e3v = ep[3];
            #pragma unroll
            for (int c = 0; c < 32; ++c) {
                const float v = e0v*cst[c] + e1v*cst[32+c] + e2v*cst[64+c] + e3v*cst[96+c];
                tile[m*HID + c] = f2bf(v);
            }
        } else {
            const int a = A_tab[eb];
            #pragma unroll
            for (int c = 0; c < 32; ++c)
                tile[m*HID + 32 + c] = f2bf(cst[128 + a*32 + c]);
        }
        __builtin_amdgcn_wave_barrier();
        { // e0 -> bufE (coalesced 16B stores)
            const uint4* src = reinterpret_cast<const uint4*>(tile);
            uint4* dst = reinterpret_cast<uint4*>(bufE + (size_t)tIdx*1024);
            #pragma unroll
            for (int qo = 0; qo < 4; ++qo) dst[lane + 32*qo] = src[lane + 32*qo];
        }
    } else {
        const size_t base = (size_t)tIdx*1024 + (size_t)m*HID + cH;
        const v8bf* pe = reinterpret_cast<const v8bf*>(bufE + base);
        const v8bf* pt = reinterpret_cast<const v8bf*>(bufT + base);
        v8bf*       pw = reinterpret_cast<v8bf*>(bufE + base);
        #pragma unroll
        for (int qo = 0; qo < 4; ++qo) {
            const v8bf ve = pe[qo];
            const v8bf vt = pt[qo];
            v8bf vo = ve;
            #pragma unroll
            for (int e = 0; e < 8; ++e) {
                const int c = cH + qo*8 + e;
                const float bnv = cst[c]*(bf2f(vt[e]) - cst[128+c])*cst[192+c] + cst[64+c];
                vo[e] = f2bf(bf2f(ve[e]) + fmaxf(bnv, 0.f));
            }
            *reinterpret_cast<v8bf*>(&tile[m*HID + cH + qo*8]) = vo;
            pw[qo] = vo;   // e1 back to bufE for the final gather
        }
    }
    __builtin_amdgcn_wave_barrier();

    // -------- phase 2: (16x64) @ (64x64) via v_wmma_f32_16x16x32_bf16 ------
    // A-frag: lane holds row M=lane&15; K sub-lanes {0..7,16..23} (lane<16) or
    // {8..15,24..31} (lane>=16) per 32-K half -> two contiguous v8bf LDS loads.
    const int mA = lane & 15;
    const int sA = (lane & 16) ? 8 : 0;
    v16bf afr[2];
    #pragma unroll
    for (int h = 0; h < 2; ++h) {
        const v8bf lo = *reinterpret_cast<const v8bf*>(&tile[mA*HID + 32*h + sA]);
        const v8bf hi = *reinterpret_cast<const v8bf*>(&tile[mA*HID + 32*h + sA + 16]);
        afr[h] = __builtin_shufflevector(lo, hi, 0,1,2,3,4,5,6,7,8,9,10,11,12,13,14,15);
    }
    v8f acc[4];
    #pragma unroll
    for (int t = 0; t < 4; ++t) {
        v8f c8 = {0.f,0.f,0.f,0.f,0.f,0.f,0.f,0.f};
        #pragma unroll
        for (int h = 0; h < 2; ++h) {
            const v16bf bfr = *reinterpret_cast<const v16bf*>(
                Upk + ((size_t)((h*4 + t)*32 + lane))*16);
            c8 = __builtin_amdgcn_wmma_f32_16x16x32_bf16(
                     false, afr[h], false, bfr, (short)0, c8, false, false);
        }
        acc[t] = c8;
    }

    // -------- phase 2b: + VxB[b,i] + VxB[b,j]  (carries eU_b) --------------
    const int nL  = lane & 15;
    const int hi8 = (lane & 16) ? 8 : 0;
    const float* vri = VxB + ((size_t)(b*NN + i))*HID;
    #pragma unroll
    for (int t = 0; t < 4; ++t) {
        const int n = t*16 + nL;
        const float vi = vri[n];
        #pragma unroll
        for (int r = 0; r < 8; ++r) {
            const int j = j0 + r + hi8;
            acc[t][r] += vi + VxB[((size_t)(b*NN + j))*HID + n];
        }
    }

    // -------- phase 3: BN partial stats + store etmp (bf16) ----------------
    __bf16* ct = ctile[wave];
    float S[4], Q[4];
    #pragma unroll
    for (int t = 0; t < 4; ++t) {
        float s = 0.f, qq = 0.f;
        #pragma unroll
        for (int r = 0; r < 8; ++r) {
            const float v = acc[t][r];
            s += v; qq += v*v;
            ct[(r + hi8)*HID + t*16 + nL] = f2bf(v);
        }
        S[t] = s; Q[t] = qq;
    }
    #pragma unroll
    for (int t = 0; t < 4; ++t) {          // fold lane<->lane^16 (same column)
        S[t] += __shfl_xor(S[t], 16);
        Q[t] += __shfl_xor(Q[t], 16);
    }
    if (lane < 16) {
        #pragma unroll
        for (int t = 0; t < 4; ++t) {
            sS[wave][t*16 + lane] = S[t];
            sQ[wave][t*16 + lane] = Q[t];
        }
    }
    __builtin_amdgcn_wave_barrier();
    { // etmp -> bufT (coalesced)
        const uint4* src = reinterpret_cast<const uint4*>(ct);
        uint4* dst = reinterpret_cast<uint4*>(bufT + (size_t)tIdx*1024);
        #pragma unroll
        for (int qo = 0; qo < 4; ++qo) dst[lane + 32*qo] = src[lane + 32*qo];
    }
    __syncthreads();
    if (threadIdx.x < 64) {                // deterministic per-block partials
        const int c = threadIdx.x;
        float s = 0.f, qq = 0.f;
        #pragma unroll
        for (int w = 0; w < 8; ++w) { s += sS[w][c]; qq += sQ[w][c]; }
        partial[(size_t)blockIdx.x*128 + c]      = s;
        partial[(size_t)blockIdx.x*128 + 64 + c] = qq;
    }
}

// ---------------------------------------------------------------------------
// Finalize BN stats: mu = S/ROWS, rsig = rsqrt(Q/ROWS - mu^2 + 1e-5)
// ---------------------------------------------------------------------------
__global__ __launch_bounds__(64) void k_stats(const float* __restrict__ partial,
                                              float* __restrict__ mu,
                                              float* __restrict__ rsig)
{
    const int c = threadIdx.x;
    float s = 0.f, q = 0.f;
    for (int blk = 0; blk < GCN_BLOCKS; ++blk) {
        s += partial[(size_t)blk*128 + c];
        q += partial[(size_t)blk*128 + 64 + c];
    }
    const float m = s / (float)ROWS;
    const float v = q / (float)ROWS - m*m;
    mu[c]   = m;
    rsig[c] = rsqrtf(v + 1e-5f);
}

// ---------------------------------------------------------------------------
// Gather 256 routed edges, apply bn2+residual, project with W_el, concat
// positional encoding + time embedding -> R (B,L,64).
// ---------------------------------------------------------------------------
__global__ __launch_bounds__(256) void k_build_R(
    const float* __restrict__ f, const int* __restrict__ route,
    const __bf16* __restrict__ bufE, const __bf16* __restrict__ bufT,
    const float* __restrict__ bn_g2, const float* __restrict__ bn_b2,
    const float* __restrict__ musig2,
    const float* __restrict__ W_el, const float* __restrict__ b_el,
    const float* __restrict__ time_embed,
    float* __restrict__ R)
{
    const int tid = threadIdx.x;          // 256 = B*L
    const int b = tid >> 6, l = tid & 63;
    const int ii = route[(b*LLEN + l)*2 + 0];
    const int jj = route[(b*LLEN + l)*2 + 1];
    const size_t row = ((size_t)(b*NN + ii))*NN + (size_t)jj;
    const __bf16* e1 = bufE + row*HID;
    const __bf16* t2 = bufT + row*HID;

    float acc[EODX];
    #pragma unroll
    for (int o = 0; o < EODX; ++o) acc[o] = b_el[o];
    for (int c = 0; c < HID; ++c) {
        const float bnv = bn_g2[c]*(bf2f(t2[c]) - musig2[c])*musig2[64+c] + bn_b2[c];
        const float e2  = bf2f(e1[c]) + fmaxf(bnv, 0.f);
        #pragma unroll
        for (int o = 0; o < EODX; ++o) acc[o] += e2 * W_el[c*EODX + o];
    }
    float* Rr = R + (size_t)tid*64;
    #pragma unroll
    for (int o = 0; o < EODX; ++o) Rr[o] = acc[o];
    #pragma unroll
    for (int mi = 0; mi < 14; ++mi) {     // pe: ang = l / 10000^(mi/7)
        const float ang = (float)l / powf(10000.f, (float)mi/7.f);
        Rr[28 + 2*mi]     = sinf(ang);
        Rr[28 + 2*mi + 1] = cosf(ang);
    }
    const int hour = (int)f[b*9 + 1];
    #pragma unroll
    for (int o = 0; o < 8; ++o) Rr[56 + o] = time_embed[hour*8 + o];
}

// q/k/v projections: one block per R row, thread per output channel.
__global__ __launch_bounds__(64) void k_qkv(
    const float* __restrict__ R,
    const float* __restrict__ Wq, const float* __restrict__ bq,
    const float* __restrict__ Wk, const float* __restrict__ bk,
    const float* __restrict__ Wv, const float* __restrict__ bv,
    float* __restrict__ q, float* __restrict__ k, float* __restrict__ v)
{
    __shared__ float rs[64];
    const int row = blockIdx.x;           // b*64+l
    const int c = threadIdx.x;
    rs[c] = R[(size_t)row*64 + c];
    __syncthreads();
    float aq = bq[c], ak = bk[c], av = bv[c];
    for (int kk = 0; kk < 64; ++kk) {
        const float rv = rs[kk];
        aq += rv*Wq[kk*64 + c];
        ak += rv*Wk[kk*64 + c];
        av += rv*Wv[kk*64 + c];
    }
    const int b = row >> 6, l = row & 63, h = c >> 4, d = c & 15;
    const size_t o = ((size_t)(b*4 + h)*64 + l)*16 + d;   // (b,h,l,d)
    q[o] = aq; k[o] = ak; v[o] = av;
}

// scores + masked softmax + context. One block per (b,h,q-row).
__global__ __launch_bounds__(64) void k_attn(
    const float* __restrict__ q, const float* __restrict__ k,
    const float* __restrict__ v, const int* __restrict__ mask,
    float* __restrict__ ctx)
{
    __shared__ float p[64];
    __shared__ float red[64];
    const int bh = blockIdx.x >> 6;       // b*4+h
    const int ql = blockIdx.x & 63;
    const int b  = bh >> 2;
    const int kk = threadIdx.x;
    const float* qr = q + ((size_t)bh*64 + ql)*16;
    const float* kr = k + ((size_t)bh*64 + kk)*16;
    float s = 0.f;
    #pragma unroll
    for (int d = 0; d < 16; ++d) s += qr[d]*kr[d];
    s *= 0.25f;                           // 1/sqrt(16)
    if (mask[(b*LLEN + ql)*LLEN + kk] == 0) s = 1e-8f;
    red[kk] = s; __syncthreads();
    for (int off = 32; off >= 1; off >>= 1) {
        if (kk < off) red[kk] = fmaxf(red[kk], red[kk + off]);
        __syncthreads();
    }
    const float mx = red[0]; __syncthreads();
    const float e = expf(s - mx);
    p[kk] = e; red[kk] = e; __syncthreads();
    for (int off = 32; off >= 1; off >>= 1) {
        if (kk < off) red[kk] += red[kk + off];
        __syncthreads();
    }
    const float inv = 1.f / red[0];
    if (kk < 16) {
        float a = 0.f;
        for (int t = 0; t < 64; ++t) a += p[t] * v[((size_t)bh*64 + t)*16 + kk];
        const int h = bh & 3;             // ctx laid out (b,l,h*16+d)
        ctx[((size_t)(b*LLEN + ql))*64 + h*16 + kk] = a * inv;
    }
}

// final projection + output assembly (B,4104)
__global__ __launch_bounds__(64) void k_out(
    const float* __restrict__ ctx, const float* __restrict__ Wd,
    const float* __restrict__ bd, const float* __restrict__ f,
    float* __restrict__ out)
{
    __shared__ float cs[64];
    const int row = blockIdx.x;           // b*64+l
    const int c = threadIdx.x;
    cs[c] = ctx[(size_t)row*64 + c];
    __syncthreads();
    float a = bd[c];
    for (int kq = 0; kq < 64; ++kq) a += cs[kq]*Wd[kq*64 + c];
    const int b = row >> 6, l = row & 63;
    out[(size_t)b*4104 + l*64 + c] = a;
    if (l == 0 && c < 8) out[(size_t)b*4104 + 4096 + c] = f[b*9 + 1 + c];
}

// ---------------------------------------------------------------------------
extern "C" void kernel_launch(void* const* d_in, const int* in_sizes, int n_in,
                              void* d_out, int out_size, void* d_ws, size_t ws_size,
                              hipStream_t stream)
{
    (void)in_sizes; (void)n_in; (void)out_size; (void)ws_size;
    // Input order: top-level dict insertion order; params flattened as a JAX
    // pytree (dict keys sorted, list in order).
    const float* f         = (const float*)d_in[0];
    const float* edge_tab  = (const float*)d_in[1];
    const float* node_tab  = (const float*)d_in[2];
    const int*   route     = (const int*)  d_in[3];
    const int*   mask      = (const int*)  d_in[4];
    const int*   A_tab     = (const int*)  d_in[5];
    const float* W_el      = (const float*)d_in[6];
    const float* W_eval    = (const float*)d_in[7];
    const float* W_node    = (const float*)d_in[8];
    const float* Wd        = (const float*)d_in[9];
    // d_in[10]=Wk, [11]=Wq, [12]=Wv (sorted: Wd,Wk,Wq,Wv)
    const float* Wk        = (const float*)d_in[10];
    const float* Wq        = (const float*)d_in[11];
    const float* Wv        = (const float*)d_in[12];
    const float* b_el      = (const float*)d_in[13];
    const float* bd        = (const float*)d_in[14];
    const float* bk        = (const float*)d_in[15];
    const float* bq        = (const float*)d_in[16];
    const float* bv        = (const float*)d_in[17];
    const float* emb_edges = (const float*)d_in[18];
    // gcn[0]: 19 bn_e_b, 20 bn_e_g, 21 bn_x_b, 22 bn_x_g, 23 eU_b, 24 eU_w,
    //         25 eV_b, 26 eV_w, 27 nU_b, 28 nU_w, 29 nV_b, 30 nV_w
    const float* g0_bn_e_b = (const float*)d_in[19];
    const float* g0_bn_e_g = (const float*)d_in[20];
    const float* g0_eU_b   = (const float*)d_in[23];
    const float* g0_eU_w   = (const float*)d_in[24];
    const float* g0_eV_b   = (const float*)d_in[25];
    const float* g0_eV_w   = (const float*)d_in[26];
    // gcn[1]: 31..42 same pattern
    const float* g1_bn_e_b = (const float*)d_in[31];
    const float* g1_bn_e_g = (const float*)d_in[32];
    const float* g1_eU_b   = (const float*)d_in[35];
    const float* g1_eU_w   = (const float*)d_in[36];
    const float* g1_eV_b   = (const float*)d_in[37];
    const float* g1_eV_w   = (const float*)d_in[38];
    const float* time_embed= (const float*)d_in[43];
    float* out = (float*)d_out;

    char* ws = (char*)d_ws;
    size_t off = 0;
    auto take = [&](size_t bytes) -> char* {
        char* p = ws + off;
        off += (bytes + 255) & ~(size_t)255;
        return p;
    };
    __bf16* bufE   = (__bf16*)take((size_t)ROWS*HID*sizeof(__bf16)); // 32MB
    __bf16* bufT   = (__bf16*)take((size_t)ROWS*HID*sizeof(__bf16)); // 32MB
    __bf16* U1pk   = (__bf16*)take(4096*sizeof(__bf16));
    __bf16* U2pk   = (__bf16*)take(4096*sizeof(__bf16));
    float*  VxB1   = (float*) take((size_t)NB*NN*HID*sizeof(float));
    float*  VxB2   = (float*) take((size_t)NB*NN*HID*sizeof(float));
    float*  partial= (float*) take((size_t)GCN_BLOCKS*128*sizeof(float));
    float*  musig  = (float*) take(256*sizeof(float)); // mu1,rsig1,mu2,rsig2
    float*  Rbuf   = (float*) take((size_t)NB*LLEN*64*sizeof(float));
    float*  qbuf   = (float*) take((size_t)NB*LLEN*64*sizeof(float));
    float*  kbuf   = (float*) take((size_t)NB*LLEN*64*sizeof(float));
    float*  vbuf   = (float*) take((size_t)NB*LLEN*64*sizeof(float));
    float*  cbuf   = (float*) take((size_t)NB*LLEN*64*sizeof(float));

    k_prep_w <<<1, 256, 0, stream>>>(g0_eU_w, g1_eU_w, U1pk, U2pk);
    k_prep_vx<<<NB*NN, 64, 0, stream>>>(f, node_tab, W_node,
                                        g0_eV_w, g0_eV_b, g0_eU_b,
                                        g1_eV_w, g1_eV_b, g1_eU_b,
                                        VxB1, VxB2);
    k_gcn<1><<<GCN_BLOCKS, 256, 0, stream>>>(f, edge_tab, A_tab, W_eval, emb_edges,
                                             U1pk, VxB1, nullptr, nullptr, nullptr,
                                             bufE, bufT, partial);
    k_stats <<<1, 64, 0, stream>>>(partial, musig, musig + 64);
    k_gcn<2><<<GCN_BLOCKS, 256, 0, stream>>>(f, nullptr, nullptr, nullptr, nullptr,
                                             U2pk, VxB2, g0_bn_e_g, g0_bn_e_b, musig,
                                             bufE, bufT, partial);
    k_stats <<<1, 64, 0, stream>>>(partial, musig + 128, musig + 192);
    k_build_R<<<1, 256, 0, stream>>>(f, route, bufE, bufT,
                                     g1_bn_e_g, g1_bn_e_b, musig + 128,
                                     W_el, b_el, time_embed, Rbuf);
    k_qkv <<<NB*LLEN, 64, 0, stream>>>(Rbuf, Wq, bq, Wk, bk, Wv, bv,
                                       qbuf, kbuf, vbuf);
    k_attn<<<NB*4*LLEN, 64, 0, stream>>>(qbuf, kbuf, vbuf, mask, cbuf);
    k_out <<<NB*LLEN, 64, 0, stream>>>(cbuf, Wd, bd, f, out);
}
